// Mamba2Block1D_53987738910741
// MI455X (gfx1250) — matrix operand order repaired and checked by reading, not compiled, not run on hardware
//
#include <hip/hip_runtime.h>
#include <hip/hip_bf16.h>

typedef __bf16 bf16_t;
typedef __attribute__((ext_vector_type(16))) __bf16 v16bf;
typedef __attribute__((ext_vector_type(8)))  float  v8f;

union AF { uint4 u[2]; v16bf v; };

#define B_   8
#define T_   2048
#define D_   1024
#define M_   (B_ * T_)        // 16384 rows

// ---------------------------------------------------------------------------
// f32 -> bf16 cast (RNE via clang fptrunc)
// ---------------------------------------------------------------------------
__global__ void cast_f32_bf16(const float* __restrict__ src,
                              bf16_t* __restrict__ dst, int n) {
    int i = blockIdx.x * blockDim.x + threadIdx.x;
    if (i < n) dst[i] = (bf16_t)src[i];
}

// ---------------------------------------------------------------------------
// WMMA GEMM: C[M,N] = A[M,K](bf16,row) * Bw[N,K](bf16,row)^T (+ addend)
// wave32; one wave computes a 64x64 tile; 16 v_wmma_f32_16x16x32_bf16 / K-step
// Fragments loaded straight from global per ISA 7.12.2 VGPR layouts.
// ---------------------------------------------------------------------------
__global__ __launch_bounds__(256) void wmma_gemm_bf16(
    const bf16_t* __restrict__ A,
    const bf16_t* __restrict__ Bw,
    const float*  __restrict__ addend,   // optional M*N residual (x), may be null
    float*        __restrict__ C,
    int M, int N, int K)
{
    const int lane  = threadIdx.x & 31;
    const int wave  = threadIdx.x >> 5;
    const int tilesN = N >> 6;
    const int tile  = blockIdx.x * 8 + wave;
    const int m0    = (tile / tilesN) * 64;
    const int n0    = (tile % tilesN) * 64;

    const int half = lane >> 4;   // 0: lanes 0-15, 1: lanes 16-31
    const int l16  = lane & 15;

    v8f acc[4][4];
    #pragma unroll
    for (int i = 0; i < 4; ++i)
        #pragma unroll
        for (int j = 0; j < 4; ++j)
            #pragma unroll
            for (int r = 0; r < 8; ++r) acc[i][j][r] = 0.0f;

    for (int k0 = 0; k0 < K; k0 += 32) {
        AF a[4], b[4];
        // A fragments: 16x32 bf16, lane l -> row l&15, K chunks [8h..8h+7],[16+8h..]
        #pragma unroll
        for (int mi = 0; mi < 4; ++mi) {
            const bf16_t* pa = A + (size_t)(m0 + mi * 16 + l16) * K + k0 + half * 8;
            a[mi].u[0] = *reinterpret_cast<const uint4*>(pa);
            a[mi].u[1] = *reinterpret_cast<const uint4*>(pa + 16);
        }
        // B fragments: 32x16; lane l -> column n (l&15), K chunks [8h..8h+7],[16+8h..]
        // B[k][n] == W[n][k] => two 16B chunks, 32B apart, in W's row n
        #pragma unroll
        for (int nj = 0; nj < 4; ++nj) {
            const bf16_t* pb = Bw + (size_t)(n0 + nj * 16 + l16) * K + k0 + half * 8;
            b[nj].u[0] = *reinterpret_cast<const uint4*>(pb);
            b[nj].u[1] = *reinterpret_cast<const uint4*>(pb + 16);
        }
        #pragma unroll
        for (int mi = 0; mi < 4; ++mi)
            #pragma unroll
            for (int nj = 0; nj < 4; ++nj)
                acc[mi][nj] = __builtin_amdgcn_wmma_f32_16x16x32_bf16(
                    false, a[mi].v, false, b[nj].v,
                    (short)0, acc[mi][nj], false, false);
    }

    // C/D layout: lanes 0-15 -> M = r, N = lane; lanes 16-31 -> M = 8+r, N = lane-16
    #pragma unroll
    for (int mi = 0; mi < 4; ++mi) {
        #pragma unroll
        for (int nj = 0; nj < 4; ++nj) {
            const int col = n0 + nj * 16 + l16;
            #pragma unroll
            for (int r = 0; r < 8; ++r) {
                const int row = m0 + mi * 16 + half * 8 + r;
                const size_t idx = (size_t)row * N + col;
                float v = acc[mi][nj][r];
                if (addend) v += addend[idx];
                C[idx] = v;
            }
        }
    }
}

// ---------------------------------------------------------------------------
// Fused sigmoid(gate) * (depthwise conv(val) + bias); writes gv over gate slot.
// proj layout: row (b*T+t) of 2048 floats: [0,1024)=gate, [1024,2048)=val
// ---------------------------------------------------------------------------
__global__ void conv_gate_kernel(float* __restrict__ proj,
                                 const float* __restrict__ conv_w,
                                 const float* __restrict__ conv_b) {
    size_t idx = (size_t)blockIdx.x * blockDim.x + threadIdx.x;   // B*T*D
    const int d  = (int)(idx & 1023);
    const size_t bt = idx >> 10;              // b*T + t
    const int t  = (int)(bt & 2047);

    float acc = conv_b[d];
    #pragma unroll
    for (int k = 0; k < 5; ++k) {
        const int tt = t + k - 2;
        if (tt >= 0 && tt < T_) {
            acc += conv_w[d * 5 + k] * proj[(bt + (size_t)(k - 2)) * 2048 + 1024 + d];
        }
    }
    const size_t gi = bt * 2048 + d;
    const float g = 1.0f / (1.0f + __expf(-proj[gi]));
    proj[gi] = g * acc;   // gv in place of gate
}

// ---------------------------------------------------------------------------
// Linear scan h = tanh(a_d)*h + gv along T; one thread per (b,d); emits bf16 y
// ---------------------------------------------------------------------------
__global__ void scan_kernel(const float* __restrict__ gv,     // proj base (gate slot)
                            const float* __restrict__ a_param,
                            bf16_t* __restrict__ yb) {
    const int idx = blockIdx.x * blockDim.x + threadIdx.x;    // B*D = 8192
    const int d = idx & 1023;
    const int b = idx >> 10;
    const float a = tanhf(a_param[d]);
    float h = 0.0f;
    const size_t base  = (size_t)b * T_ * 2048 + d;
    const size_t ybase = (size_t)b * T_ * D_   + d;
    for (int t = 0; t < T_; ++t) {
        h = fmaf(a, h, gv[base + (size_t)t * 2048]);
        yb[ybase + (size_t)t * D_] = (bf16_t)h;
    }
}

// ---------------------------------------------------------------------------
// LayerNorm over D=1024; one 256-thread block per row, float4 per thread
// ---------------------------------------------------------------------------
__global__ __launch_bounds__(256) void ln_kernel(const float* __restrict__ res,
                                                 const float* __restrict__ w,
                                                 const float* __restrict__ bb,
                                                 float* __restrict__ out) {
    __shared__ float s_sum[8], s_sq[8];
    const size_t row = blockIdx.x;
    const float4 v = reinterpret_cast<const float4*>(res + row * D_)[threadIdx.x];
    float sum = v.x + v.y + v.z + v.w;
    float sq  = v.x * v.x + v.y * v.y + v.z * v.z + v.w * v.w;
    #pragma unroll
    for (int off = 16; off; off >>= 1) {
        sum += __shfl_xor(sum, off);
        sq  += __shfl_xor(sq,  off);
    }
    const int wv = threadIdx.x >> 5, lane = threadIdx.x & 31;
    if (lane == 0) { s_sum[wv] = sum; s_sq[wv] = sq; }
    __syncthreads();
    if (wv == 0) {
        float a = (lane < 8) ? s_sum[lane] : 0.0f;
        float q = (lane < 8) ? s_sq[lane]  : 0.0f;
        #pragma unroll
        for (int off = 4; off; off >>= 1) {
            a += __shfl_xor(a, off);
            q += __shfl_xor(q, off);
        }
        if (lane == 0) { s_sum[0] = a; s_sq[0] = q; }
    }
    __syncthreads();
    const float mean = s_sum[0] * (1.0f / D_);
    const float var  = s_sq[0]  * (1.0f / D_) - mean * mean;
    const float inv  = rsqrtf(var + 1e-5f);
    const float4 ww = reinterpret_cast<const float4*>(w)[threadIdx.x];
    const float4 bv = reinterpret_cast<const float4*>(bb)[threadIdx.x];
    float4 o;
    o.x = (v.x - mean) * inv * ww.x + bv.x;
    o.y = (v.y - mean) * inv * ww.y + bv.y;
    o.z = (v.z - mean) * inv * ww.z + bv.z;
    o.w = (v.w - mean) * inv * ww.w + bv.w;
    reinterpret_cast<float4*>(out + row * D_)[threadIdx.x] = o;
}

// ---------------------------------------------------------------------------
extern "C" void kernel_launch(void* const* d_in, const int* in_sizes, int n_in,
                              void* d_out, int out_size, void* d_ws, size_t ws_size,
                              hipStream_t stream) {
    const float* x       = (const float*)d_in[0];
    const float* W_in    = (const float*)d_in[1];
    const float* conv_w  = (const float*)d_in[2];
    const float* conv_b  = (const float*)d_in[3];
    const float* a_param = (const float*)d_in[4];
    const float* W_out   = (const float*)d_in[5];
    const float* ln_w    = (const float*)d_in[6];
    const float* ln_b    = (const float*)d_in[7];
    float* out = (float*)d_out;

    char* ws = (char*)d_ws;
    const size_t OFF_XB    = 0;                              // 16384*1024 bf16 = 32MB
    const size_t OFF_WINB  = OFF_XB    + (size_t)M_ * D_ * 2;        // 4MB
    const size_t OFF_WOUTB = OFF_WINB  + (size_t)2 * D_ * D_ * 2;    // 2MB
    const size_t OFF_PROJ  = OFF_WOUTB + (size_t)D_ * D_ * 2;        // 128MB
    const size_t OFF_YB    = OFF_PROJ  + (size_t)M_ * 2 * D_ * 4;    // 32MB
    const size_t OFF_RES   = OFF_YB    + (size_t)M_ * D_ * 2;        // 64MB

    bf16_t* xb    = (bf16_t*)(ws + OFF_XB);
    bf16_t* winb  = (bf16_t*)(ws + OFF_WINB);
    bf16_t* woutb = (bf16_t*)(ws + OFF_WOUTB);
    float*  proj  = (float*) (ws + OFF_PROJ);
    bf16_t* yb    = (bf16_t*)(ws + OFF_YB);
    float*  res   = (float*) (ws + OFF_RES);

    // 1. casts to bf16
    cast_f32_bf16<<<(M_ * D_ + 255) / 256, 256, 0, stream>>>(x, xb, M_ * D_);
    cast_f32_bf16<<<(2 * D_ * D_ + 255) / 256, 256, 0, stream>>>(W_in, winb, 2 * D_ * D_);
    cast_f32_bf16<<<(D_ * D_ + 255) / 256, 256, 0, stream>>>(W_out, woutb, D_ * D_);

    // 2. proj = x @ W_in^T   (M=16384, N=2048, K=1024) -> 8192 tiles / 8 waves
    wmma_gemm_bf16<<<(M_ / 64) * (2 * D_ / 64) / 8, 256, 0, stream>>>(
        xb, winb, nullptr, proj, M_, 2 * D_, D_);

    // 3. gv = sigmoid(gate) * (depthwise_conv(val) + b), in place over gate
    conv_gate_kernel<<<(size_t)M_ * D_ / 256, 256, 0, stream>>>(proj, conv_w, conv_b);

    // 4. scan along T, emit y as bf16
    scan_kernel<<<(B_ * D_) / 256, 256, 0, stream>>>(proj, a_param, yb);

    // 5. res = x + y @ W_out^T   (M=16384, N=1024, K=1024)
    wmma_gemm_bf16<<<(M_ / 64) * (D_ / 64) / 8, 256, 0, stream>>>(
        yb, woutb, x, res, M_, D_, D_);

    // 6. LayerNorm
    ln_kernel<<<M_, 256, 0, stream>>>(res, ln_w, ln_b, out);
}
